// CausalMultiheadSelfAttention_45226005627460
// MI455X (gfx1250) — matrix operand order repaired and checked by reading, not compiled, not run on hardware
//
#include <hip/hip_runtime.h>

// ---------------- problem constants ----------------
#define BS   4
#define SL   2048
#define DM   1024
#define NH   16
#define HD   64
#define MTOT (BS*SL)   // 8192 rows
#define NQKV (3*DM)    // 3072 qkv output cols

typedef unsigned short u16;
typedef __attribute__((ext_vector_type(16))) __bf16 v16bf;
typedef __attribute__((ext_vector_type(8)))  float  v8f;
typedef __attribute__((ext_vector_type(8)))  u16    v8u;
typedef __attribute__((ext_vector_type(16))) u16    v16u;
typedef __attribute__((ext_vector_type(4)))  u16    v4u;
typedef __attribute__((ext_vector_type(4)))  float  v4f;

// fp32 -> bf16 round-to-nearest-even
static __device__ __forceinline__ u16 f2bf(float x) {
  unsigned u = __builtin_bit_cast(unsigned, x);
  u += 0x7FFFu + ((u >> 16) & 1u);
  return (u16)(u >> 16);
}

static __device__ __forceinline__ v8f vzero8() {
  v8f v;
#pragma unroll
  for (int i = 0; i < 8; ++i) v[i] = 0.0f;
  return v;
}

// Build a 16-element bf16 fragment from two contiguous 8-element (16B) chunks.
static __device__ __forceinline__ v16bf ldfrag(const u16* p0, const u16* p1) {
  v8u a = *(const v8u*)p0;
  v8u b = *(const v8u*)p1;
  v16u c = __builtin_shufflevector(a, b, 0,1,2,3,4,5,6,7,8,9,10,11,12,13,14,15);
  return __builtin_bit_cast(v16bf, c);
}

static __device__ __forceinline__ v8f wmma_bf16(v16bf a, v16bf b, v8f c) {
  return __builtin_amdgcn_wmma_f32_16x16x32_bf16(false, a, false, b, (short)0, c,
                                                 false, false);
}

// ---------------- conversion kernels ----------------
__global__ __launch_bounds__(256) void k_cvt_x(const float* __restrict__ x,
                                               u16* __restrict__ xb, int n4) {
  int i = blockIdx.x * blockDim.x + threadIdx.x;
  if (i >= n4) return;
  v4f v = *(const v4f*)(x + (size_t)i * 4);
  v4u o;
  o[0] = f2bf(v[0]); o[1] = f2bf(v[1]); o[2] = f2bf(v[2]); o[3] = f2bf(v[3]);
  *(v4u*)(xb + (size_t)i * 4) = o;
}

// w is [K][N] fp32 ; wt is [N][K] bf16 (transposed so B-fragments are contiguous)
__global__ __launch_bounds__(256) void k_cvt_wT(const float* __restrict__ w,
                                                u16* __restrict__ wt, int K, int N) {
  int i = blockIdx.x * blockDim.x + threadIdx.x;  // i = n*K + k
  if (i >= K * N) return;
  int n = i / K, k = i - n * K;
  wt[i] = f2bf(w[(size_t)k * N + n]);
}

// ---------------- shared 32x64 GEMM core (ping-pong double buffer) ----------------
// A rows at arow0/arow1 (already offset by lane), B rows at brow; K = DM.
// On return acc[ms][t] holds the 16x16 C tiles.
static __device__ __forceinline__ void gemm_core_32x64(const u16* __restrict__ arow0,
                                                       const u16* __restrict__ arow1,
                                                       const u16* __restrict__ brow,
                                                       v8f acc[2][4]) {
#pragma unroll
  for (int ms = 0; ms < 2; ++ms)
#pragma unroll
    for (int t = 0; t < 4; ++t) acc[ms][t] = vzero8();

  // prime buffer A with k = 0
  v16bf a0A = ldfrag(arow0, arow0 + 16);
  v16bf a1A = ldfrag(arow1, arow1 + 16);
  v16bf bA[4];
#pragma unroll
  for (int t = 0; t < 4; ++t) {
    const u16* bp = brow + (size_t)t * 16 * DM;
    bA[t] = ldfrag(bp, bp + 16);
  }

  for (int k0 = 0; k0 < DM; k0 += 64) {
    // load buffer B at k0+32 (overlaps with buffer-A WMMAs)
    const int kb_ = k0 + 32;
    v16bf a0B = ldfrag(arow0 + kb_, arow0 + kb_ + 16);
    v16bf a1B = ldfrag(arow1 + kb_, arow1 + kb_ + 16);
    v16bf bB[4];
#pragma unroll
    for (int t = 0; t < 4; ++t) {
      const u16* bp = brow + (size_t)t * 16 * DM + kb_;
      bB[t] = ldfrag(bp, bp + 16);
    }
    // consume buffer A (k0)
#pragma unroll
    for (int t = 0; t < 4; ++t) {
      acc[0][t] = wmma_bf16(a0A, bA[t], acc[0][t]);
      acc[1][t] = wmma_bf16(a1A, bA[t], acc[1][t]);
    }
    // load buffer A at k0+64 (branchless wrap on last iteration)
    const int ka_ = (k0 + 64) & (DM - 1);
    a0A = ldfrag(arow0 + ka_, arow0 + ka_ + 16);
    a1A = ldfrag(arow1 + ka_, arow1 + ka_ + 16);
#pragma unroll
    for (int t = 0; t < 4; ++t) {
      const u16* bp = brow + (size_t)t * 16 * DM + ka_;
      bA[t] = ldfrag(bp, bp + 16);
    }
    // consume buffer B (k0+32)
#pragma unroll
    for (int t = 0; t < 4; ++t) {
      acc[0][t] = wmma_bf16(a0B, bB[t], acc[0][t]);
      acc[1][t] = wmma_bf16(a1B, bB[t], acc[1][t]);
    }
  }
}

// ---------------- QKV projection GEMM ----------------
// C[MTOT, NQKV] = Xb @ WqkvT^T ; per wave: 32x64 output tile.
// Q,K stored naturally (b,h,seq,hd) bf16 ; V stored transposed (b,h,hd,seq) bf16.
__global__ __launch_bounds__(256) void k_qkv_gemm(const u16* __restrict__ Xb,
                                                  const u16* __restrict__ Wt,
                                                  u16* __restrict__ Qb,
                                                  u16* __restrict__ Kb,
                                                  u16* __restrict__ Vt) {
  const int lane = threadIdx.x & 31;
  const int wid  = (blockIdx.x * blockDim.x + threadIdx.x) >> 5;
  const int NT64 = NQKV / 64;  // 48
  const int tn = wid % NT64, tm = wid / NT64;
  const int m0 = tm * 32, n0 = tn * 64;
  const int hf = lane >> 4, ln = lane & 15;

  const u16* arow0 = Xb + (size_t)(m0 + ln) * DM + 8 * hf;   // rows m0..m0+15
  const u16* arow1 = arow0 + (size_t)16 * DM;                // rows m0+16..m0+31
  const u16* brow  = Wt + (size_t)(n0 + ln) * DM + 8 * hf;

  v8f acc[2][4];
  gemm_core_32x64(arow0, arow1, brow, acc);

#pragma unroll
  for (int ms = 0; ms < 2; ++ms) {
    const int mb = m0 + ms * 16 + 8 * hf;  // C layout: VGPR r -> row mb+r, col ln
    const int batch = mb / SL, sb = mb % SL;
#pragma unroll
    for (int t = 0; t < 4; ++t) {
      const int ng = n0 + t * 16 + ln;
      const int region = ng >> 10;  // 0=Q 1=K 2=V (uniform per wave)
      const int nl = ng & 1023;
      const int head = nl >> 6, d = nl & 63;
      if (region == 2) {
        // transposed store: 8 consecutive seq values -> one b128
        v8u pk;
#pragma unroll
        for (int r = 0; r < 8; ++r) pk[r] = f2bf(acc[ms][t][r]);
        *(v8u*)(Vt + ((size_t)((batch * NH + head) * HD + d)) * SL + sb) = pk;
      } else {
        u16* dst = (region == 0 ? Qb : Kb) +
                   ((size_t)((batch * NH + head) * SL + sb)) * HD + d;
#pragma unroll
        for (int r = 0; r < 8; ++r) dst[(size_t)r * HD] = f2bf(acc[ms][t][r]);
      }
    }
  }
}

// ---------------- flash attention ----------------
// one wave per (batch, head, 16-query tile); online softmax over 32-key blocks
__global__ __launch_bounds__(32) void k_flash(const u16* __restrict__ Qb,
                                              const u16* __restrict__ Kb,
                                              const u16* __restrict__ Vt,
                                              u16* __restrict__ Ob) {
  __shared__ u16 plds[16 * 32];  // P tile, bf16, row-major [16][32]
  const int lane = threadIdx.x;
  const int hf = lane >> 4, ln = lane & 15;
  int bid = blockIdx.x;
  const int qt = bid & 127; bid >>= 7;
  const int h  = bid & 15;  const int b = bid >> 4;
  const int q0 = qt * 16;

  const u16* Qh = Qb + (size_t)((b * NH + h) * SL) * HD;
  const u16* Kh = Kb + (size_t)((b * NH + h) * SL) * HD;
  const u16* Vh = Vt + (size_t)((b * NH + h) * HD) * SL;

  // Q A-fragments for head_dim k-steps 0 and 32 (reused every iteration)
  v16bf qa0, qa1;
  {
    const u16* p = Qh + (size_t)(q0 + ln) * HD + 16 * hf;
    qa0 = ldfrag(p,      p + 8);
    qa1 = ldfrag(p + 32, p + 40);
  }

  v8f o[4] = {vzero8(), vzero8(), vzero8(), vzero8()};
  float mrun[8], lrun[8];
#pragma unroll
  for (int r = 0; r < 8; ++r) { mrun[r] = -1e30f; lrun[r] = 0.0f; }

  const float SC  = 0.125f;             // 1/sqrt(64)
  const float L2E = 1.44269504088896f;  // log2(e)
  const int nkb = (q0 + 16 + 31) >> 5;  // causal: key blocks needed

  for (int kb = 0; kb < nkb * 32; kb += 32) {
    // ---- S = Q K^T (two 16x16 tiles over 32 keys, K-dim = 64) ----
    v8f s0 = vzero8(), s1 = vzero8();
    {
      const u16* kp0 = Kh + (size_t)(kb + ln) * HD + 16 * hf;
      s0 = wmma_bf16(qa0, ldfrag(kp0,      kp0 + 8),  s0);
      s0 = wmma_bf16(qa1, ldfrag(kp0 + 32, kp0 + 40), s0);
      const u16* kp1 = Kh + (size_t)(kb + 16 + ln) * HD + 16 * hf;
      s1 = wmma_bf16(qa0, ldfrag(kp1,      kp1 + 8),  s1);
      s1 = wmma_bf16(qa1, ldfrag(kp1 + 32, kp1 + 40), s1);
    }
    // issue V loads now: their latency hides under the softmax VALU work
    v16bf vf[4];
#pragma unroll
    for (int t = 0; t < 4; ++t) {
      const u16* vp = Vh + (size_t)(t * 16 + ln) * SL + kb + 16 * hf;
      vf[t] = ldfrag(vp, vp + 8);
    }
    if (kb + 64 < SL) __builtin_prefetch(Kh + (size_t)(kb + 32 + ln) * HD, 0, 1);

    // ---- causal mask + scale (C layout: row r+8*hf, keys kb+ln / kb+16+ln) ----
    float t0[8], t1[8], p0[8], p1[8];
    const int key0 = kb + ln, key1 = kb + 16 + ln;
#pragma unroll
    for (int r = 0; r < 8; ++r) {
      const int qg = q0 + r + 8 * hf;
      t0[r] = (key0 <= qg) ? s0[r] * SC : -1e30f;
      t1[r] = (key1 <= qg) ? s1[r] * SC : -1e30f;
    }

    // ---- online softmax update (row reductions across 16-lane halves) ----
#pragma unroll
    for (int r = 0; r < 8; ++r) {
      float rm = fmaxf(t0[r], t1[r]);
#pragma unroll
      for (int m = 1; m < 16; m <<= 1) rm = fmaxf(rm, __shfl_xor(rm, m, 32));
      const float mnew = fmaxf(mrun[r], rm);
      const float corr = exp2f((mrun[r] - mnew) * L2E);
      p0[r] = exp2f((t0[r] - mnew) * L2E);
      p1[r] = exp2f((t1[r] - mnew) * L2E);
      float rs = p0[r] + p1[r];
#pragma unroll
      for (int m = 1; m < 16; m <<= 1) rs += __shfl_xor(rs, m, 32);
      lrun[r] = lrun[r] * corr + rs;
      mrun[r] = mnew;
#pragma unroll
      for (int t = 0; t < 4; ++t) o[t][r] *= corr;
    }

    // ---- C-layout -> A-layout via LDS bounce (bf16) ----
#pragma unroll
    for (int r = 0; r < 8; ++r) {
      const int row = r + 8 * hf;
      plds[row * 32 + ln]      = f2bf(p0[r]);
      plds[row * 32 + 16 + ln] = f2bf(p1[r]);
    }
    __builtin_amdgcn_wave_barrier();
    v16bf pf;
    {
      const u16* pp = &plds[ln * 32 + 16 * hf];
      pf = ldfrag(pp, pp + 8);
    }
    __builtin_amdgcn_wave_barrier();

    // ---- O += P V  (V transposed [d][seq] -> contiguous B fragments) ----
#pragma unroll
    for (int t = 0; t < 4; ++t) o[t] = wmma_bf16(pf, vf[t], o[t]);
  }

  // ---- normalize, write bf16 attention output (natural [b*SL+s][DM]) ----
#pragma unroll
  for (int t = 0; t < 4; ++t) {
    const int col = h * HD + t * 16 + ln;
#pragma unroll
    for (int r = 0; r < 8; ++r) {
      const int s = q0 + r + 8 * hf;
      Ob[(size_t)(b * SL + s) * DM + col] = f2bf(o[t][r] / lrun[r]);
    }
  }
}

// ---------------- output projection GEMM ----------------
__global__ __launch_bounds__(256) void k_proj_gemm(const u16* __restrict__ Ab,
                                                   const u16* __restrict__ Wt,
                                                   float* __restrict__ out) {
  const int lane = threadIdx.x & 31;
  const int wid  = (blockIdx.x * blockDim.x + threadIdx.x) >> 5;
  const int NT64 = DM / 64;  // 16
  const int tn = wid % NT64, tm = wid / NT64;
  const int m0 = tm * 32, n0 = tn * 64;
  const int hf = lane >> 4, ln = lane & 15;

  const u16* arow0 = Ab + (size_t)(m0 + ln) * DM + 8 * hf;
  const u16* arow1 = arow0 + (size_t)16 * DM;
  const u16* brow  = Wt + (size_t)(n0 + ln) * DM + 8 * hf;

  v8f acc[2][4];
  gemm_core_32x64(arow0, arow1, brow, acc);

#pragma unroll
  for (int ms = 0; ms < 2; ++ms) {
    const int mb = m0 + ms * 16 + 8 * hf;
#pragma unroll
    for (int t = 0; t < 4; ++t) {
      const int col = n0 + t * 16 + ln;
#pragma unroll
      for (int r = 0; r < 8; ++r)
        out[(size_t)(mb + r) * DM + col] = acc[ms][t][r];
    }
  }
}

// ---------------- host launcher ----------------
extern "C" void kernel_launch(void* const* d_in, const int* in_sizes, int n_in,
                              void* d_out, int out_size, void* d_ws, size_t ws_size,
                              hipStream_t stream) {
  (void)in_sizes; (void)n_in; (void)out_size; (void)ws_size;
  const float* x      = (const float*)d_in[0];  // (4,2048,1024)
  const float* w_qkv  = (const float*)d_in[1];  // (1024,3072)
  const float* w_proj = (const float*)d_in[2];  // (1024,1024)
  float* out = (float*)d_out;

  // workspace layout (bytes, all offsets 256-aligned)
  char* ws = (char*)d_ws;
  const size_t SZ_XB  = (size_t)MTOT * DM * 2;   // 16 MiB
  const size_t SZ_WQT = (size_t)NQKV * DM * 2;   //  6 MiB
  const size_t SZ_WPT = (size_t)DM * DM * 2;     //  2 MiB
  const size_t SZ_HBF = (size_t)MTOT * DM * 2;   // 16 MiB each for Q/K/Vt/O
  u16* Xb  = (u16*)(ws);
  u16* WqT = (u16*)(ws + SZ_XB);
  u16* WpT = (u16*)(ws + SZ_XB + SZ_WQT);
  u16* Qb  = (u16*)(ws + SZ_XB + SZ_WQT + SZ_WPT);
  u16* Kb  = (u16*)(ws + SZ_XB + SZ_WQT + SZ_WPT + SZ_HBF);
  u16* Vt  = (u16*)(ws + SZ_XB + SZ_WQT + SZ_WPT + 2 * SZ_HBF);
  u16* Ob  = (u16*)(ws + SZ_XB + SZ_WQT + SZ_WPT + 3 * SZ_HBF);

  // 1) fp32 -> bf16 conversions (+ weight transposes)
  {
    int n4 = MTOT * DM / 4;
    k_cvt_x<<<(n4 + 255) / 256, 256, 0, stream>>>(x, Xb, n4);
    int nq = DM * NQKV;
    k_cvt_wT<<<(nq + 255) / 256, 256, 0, stream>>>(w_qkv, WqT, DM, NQKV);
    int np = DM * DM;
    k_cvt_wT<<<(np + 255) / 256, 256, 0, stream>>>(w_proj, WpT, DM, DM);
  }
  // 2) fused QKV projection (V written transposed per head), 32x64 tiles
  {
    int waves = (MTOT / 32) * (NQKV / 64);       // 12288
    k_qkv_gemm<<<waves * 32 / 256, 256, 0, stream>>>(Xb, WqT, Qb, Kb, Vt);
  }
  // 3) causal flash attention: one wave per (b, h, 16-query tile)
  {
    int blocks = BS * NH * (SL / 16);            // 8192
    k_flash<<<blocks, 32, 0, stream>>>(Qb, Kb, Vt, Ob);
  }
  // 4) output projection, fp32 result, 32x64 tiles
  {
    int waves = (MTOT / 32) * (DM / 64);         // 4096
    k_proj_gemm<<<waves * 32 / 256, 256, 0, stream>>>(Ob, WpT, out);
  }
}
